// TargetDiff_17308718203686
// MI455X (gfx1250) — compile-verified, matrix operand order checked by reading, not run on hardware
//
#include <hip/hip_runtime.h>
#include <hip/hip_bf16.h>

typedef __attribute__((ext_vector_type(16))) _Float16 v16h;
typedef __attribute__((ext_vector_type(8)))  _Float16 v8h;
typedef __attribute__((ext_vector_type(8)))  float    v8f;

#define BB 64     // batch
#define VV 50     // visits
#define CCODES 40 // codes per visit
#define DD 256    // embed dim
#define HH 256    // lstm hidden
#define FH 1024   // 4*H
#define GGEN 5
#define KTOP 3
#define TDIFF 1000

// ---------------------------------------------------------------------------
// deterministic counter-based RNG (stand-in for jax threefry)
// ---------------------------------------------------------------------------
__device__ inline unsigned kh(unsigned x) {
  x ^= x >> 16; x *= 0x7feb352du; x ^= x >> 15; x *= 0x846ca68bu; x ^= x >> 16;
  return x;
}
__device__ inline unsigned kkey(int g, int s, int i) {
  return kh((unsigned)(g * 131071 + s * 8191) ^ (unsigned)i * 0x9E3779B9u);
}
__device__ inline float u01f(unsigned h) {
  return ((h >> 8) + 1u) * (1.f / 16777216.f);
}
__device__ inline float gaussk(int g, int s, int i) {
  unsigned h1 = kkey(g, s, 2 * i), h2 = kkey(g, s, 2 * i + 1);
  float r = sqrtf(-2.f * logf(u01f(h1)));
  return r * cosf(6.28318530718f * u01f(h2));
}
__device__ inline int tstep(int g, int s, int b) {
  return (int)(kkey(g, s, b) % (unsigned)TDIFF);
}
__device__ inline float sigm(float x) { return 1.f / (1.f + __expf(-x)); }

// ---------------------------------------------------------------------------
// WMMA helpers — fragment layouts per CDNA5 ISA 7.12.2 (wave32).
// For 16-bit fragments each lane holds two contiguous 8-half chunks:
//   K = kb..kb+7 and kb+16..kb+23   (kb = k0 + 8*(lane>=16))
// so with K contiguous in memory a fragment is two 128-bit loads.
// ---------------------------------------------------------------------------
__device__ inline v8f wmma32(v16h a, v16h b, v8f c) {
  return __builtin_amdgcn_wmma_f32_16x16x32_f16(false, a, false, b, (short)0, c,
                                                false, false);
}
// row-major source with contiguous K: src[row*ld + k]
__device__ inline v16h fragRM(const _Float16* src, int ld, int row0, int k0, int lane) {
  int r = row0 + (lane & 15);
  int kb = k0 + ((lane >> 4) << 3);
  const _Float16* base = src + r * ld + kb;
  v8h lo = *(const v8h*)(base);
  v8h hi = *(const v8h*)(base + 16);
  return __builtin_shufflevector(lo, hi, 0, 1, 2, 3, 4, 5, 6, 7,
                                 8, 9, 10, 11, 12, 13, 14, 15);
}

// ---------------------------------------------------------------------------
// Generic WMMA GEMM: C[M,N] (f32) = op(A[M,K] f32->f16) x Wt (f16, transposed:
// Wt[n*ldw + k]) + bias.  flags: relu, accumulate, ashift (+/-1 row shift in
// VV-periodic sequences, out-of-sequence rows read as 0), cmode (epilogue
// writes C in scan layout [T][N][B] instead of [B*T][N]).
// ---------------------------------------------------------------------------
#define BM 64
#define BN 64
#define BK 32
__global__ __launch_bounds__(256) void k_gemm(
    const float* __restrict__ A, const _Float16* __restrict__ Wt,
    const float* __restrict__ bias, float* __restrict__ C,
    int M, int N, int K, int ldw, int relu, int accm, int ashift, int cmode) {
  __shared__ alignas(16) _Float16 As[BM * BK];   // [m][k]
  __shared__ alignas(16) _Float16 Bs[BN * BK];   // [n][k]  (transposed tile)
  int m0 = blockIdx.y * BM, n0 = blockIdx.x * BN;
  int tid = threadIdx.x, wv = tid >> 5, lane = tid & 31;
  int msub = wv & 3, nsub0 = (wv >> 2) * 2;
  int hi = lane >> 4, ln = lane & 15;
  v8f acc0 = {0.f, 0.f, 0.f, 0.f, 0.f, 0.f, 0.f, 0.f};
  v8f acc1 = {0.f, 0.f, 0.f, 0.f, 0.f, 0.f, 0.f, 0.f};
  for (int k0 = 0; k0 < K; k0 += BK) {
    for (int i = tid; i < BM * BK; i += 256) {
      int r = i >> 5, c = i & 31;
      int m = m0 + r;
      float v = 0.f;
      if (ashift == 0) {
        v = A[m * K + k0 + c];
      } else {
        int tt = m % VV + ashift;  // constant modulo -> mul/shift
        if (tt >= 0 && tt < VV) v = A[(m + ashift) * K + k0 + c];
      }
      As[i] = (_Float16)v;
    }
    for (int i = tid; i < BN * BK; i += 256) {
      int n = i >> 5, c = i & 31;
      Bs[i] = Wt[(size_t)(n0 + n) * ldw + k0 + c];
    }
    if (k0 + BK < K) {  // prefetch next W tile (global_prefetch_b8)
      __builtin_prefetch(Wt + (size_t)(n0 + (tid & 63)) * ldw + k0 + BK, 0, 1);
    }
    __syncthreads();
    v16h a  = fragRM(As, BK, msub * 16, 0, lane);
    v16h b0 = fragRM(Bs, BK, nsub0 * 16, 0, lane);
    v16h b1 = fragRM(Bs, BK, (nsub0 + 1) * 16, 0, lane);
    acc0 = wmma32(a, b0, acc0);
    acc1 = wmma32(a, b1, acc1);
    __syncthreads();
  }
#pragma unroll
  for (int j = 0; j < 2; ++j) {
    int col = n0 + (nsub0 + j) * 16 + ln;
    float bia = bias ? bias[col] : 0.f;
#pragma unroll
    for (int r = 0; r < 8; ++r) {
      int row = m0 + msub * 16 + r + (hi << 3);
      size_t idx;
      if (cmode) {
        int bq = row / VV, tq = row % VV;  // constant divisor
        idx = ((size_t)tq * N + col) * BB + bq;
      } else {
        idx = (size_t)row * N + col;
      }
      float v = (j == 0 ? acc0[r] : acc1[r]) + bia;
      if (accm) v += C[idx];
      if (relu) v = v > 0.f ? v : 0.f;
      C[idx] = v;
    }
  }
}

// ---------------------------------------------------------------------------
// Persistent LSTM scan: one block per direction (grid=2), 32 waves.
// pre is laid out [T][4H][B] so a lane's 8 accumulator elements (consecutive
// batch rows, fixed column) are two float4 loads. Per step: gates[64,1024] =
// pre[t] + h @ wh via 256 WMMA tiles; c stays in registers, h stays in LDS
// (f16). whT is transposed (whT[n*256+k]) so B frags are 2x b128 loads.
// ---------------------------------------------------------------------------
__global__ __launch_bounds__(1024) void k_scan(
    const float* __restrict__ preL, const float* __restrict__ preR,
    const _Float16* __restrict__ whL, const _Float16* __restrict__ whR,
    float* __restrict__ outL, float* __restrict__ outR) {
  __shared__ alignas(16) _Float16 gates[BB * FH];  // 128 KB
  __shared__ alignas(16) _Float16 hbuf[BB * HH];   //  32 KB
  const float* pre = (blockIdx.x == 0) ? preL : preR;
  const _Float16* wh = (blockIdx.x == 0) ? whL : whR;
  float* out = (blockIdx.x == 0) ? outL : outR;
  int tid = threadIdx.x, wv = tid >> 5, lane = tid & 31;
  int hi = lane >> 4, ln = lane & 15;
  int bb_act = tid & 63, cb_act = (tid >> 6) << 4;
  float creg[16];
#pragma unroll
  for (int j = 0; j < 16; ++j) creg[j] = 0.f;
  for (int i = tid; i < BB * HH; i += 1024) hbuf[i] = (_Float16)0.f;
  __syncthreads();
  for (int t = 0; t < VV; ++t) {
    const float* pret = pre + (size_t)t * FH * BB;
#pragma unroll
    for (int i = 0; i < 8; ++i) {
      int linear = wv * 8 + i;
      int mt = linear & 3, nt = linear >> 2;
      int row0 = mt << 4, col0 = nt << 4;
      // acc init: pre[t][col][b] with b contiguous -> 2x float4
      const float* pb = pret + (size_t)(col0 + ln) * BB + row0 + (hi << 3);
      float4 p0 = ((const float4*)pb)[0];
      float4 p1 = ((const float4*)pb)[1];
      v8f acc = {p0.x, p0.y, p0.z, p0.w, p1.x, p1.y, p1.z, p1.w};
#pragma unroll
      for (int k = 0; k < 8; ++k) {
        v16h a = fragRM(hbuf, HH, row0, k * 32, lane);
        v16h b = fragRM(wh, HH, col0, k * 32, lane);
        acc = wmma32(a, b, acc);
      }
#pragma unroll
      for (int r = 0; r < 8; ++r) {
        int bbr = row0 + r + (hi << 3);
        gates[bbr * FH + col0 + ln] = (_Float16)acc[r];
      }
    }
    __syncthreads();
    {
      // vectorized gate fetch: 4 gate blocks x 16 contiguous halves
      int gbase = bb_act * FH + cb_act;
      const v8h* g0 = (const v8h*)&gates[gbase];
      const v8h* g1 = (const v8h*)&gates[gbase + 256];
      const v8h* g2 = (const v8h*)&gates[gbase + 512];
      const v8h* g3 = (const v8h*)&gates[gbase + 768];
      v8h vi[2] = {g0[0], g0[1]};
      v8h vf[2] = {g1[0], g1[1]};
      v8h vg[2] = {g2[0], g2[1]};
      v8h vo[2] = {g3[0], g3[1]};
      alignas(16) _Float16 hl[16];
      alignas(16) float ho[16];
#pragma unroll
      for (int j = 0; j < 16; ++j) {
        float gi = (float)vi[j >> 3][j & 7];
        float gf = (float)vf[j >> 3][j & 7];
        float gg = (float)vg[j >> 3][j & 7];
        float go = (float)vo[j >> 3][j & 7];
        gi = sigm(gi); gf = sigm(gf); go = sigm(go); gg = tanhf(gg);
        float c = gf * creg[j] + gi * gg;
        creg[j] = c;
        float h = go * tanhf(c);
        hl[j] = (_Float16)h;
        ho[j] = h;
      }
      v8h* hp = (v8h*)&hbuf[bb_act * HH + cb_act];
      hp[0] = *(v8h*)&hl[0];
      hp[1] = *(v8h*)&hl[8];
      float4* op = (float4*)(out + (size_t)(bb_act * VV + t) * HH + cb_act);
      op[0] = *(float4*)&ho[0];
      op[1] = *(float4*)&ho[4];
      op[2] = *(float4*)&ho[8];
      op[3] = *(float4*)&ho[12];
    }
    __syncthreads();
  }
}

// ---------------------------------------------------------------------------
// small kernels
// ---------------------------------------------------------------------------
__global__ void k_cvtT(const float* __restrict__ s, _Float16* __restrict__ d,
                       int K, int N) {  // d[n*K+k] = s[k*N+n]
  int i = blockIdx.x * 256 + threadIdx.x;
  if (i >= K * N) return;
  int k = i / N, n = i % N;
  d[(size_t)n * K + k] = (_Float16)s[i];
}
__global__ void k_copy(const float* __restrict__ s, float* __restrict__ d, int n) {
  int i = blockIdx.x * 256 + threadIdx.x;
  if (i < n) d[i] = s[i];
}
__global__ void k_flip(const float* __restrict__ s, float* __restrict__ d) {
  int i = blockIdx.x * 256 + threadIdx.x;
  if (i >= BB * VV * DD) return;
  int c = i & 255, t = (i >> 8) % VV, b = i / (VV * DD);
  d[((b * VV + (VV - 1 - t)) << 8) + c] = s[i];
}
__global__ void k_alphabar(const float* __restrict__ betas, float* __restrict__ ab) {
  if (blockIdx.x == 0 && threadIdx.x == 0) {
    float p = 1.f;
    for (int i = 0; i < TDIFF; ++i) { p *= (1.f - betas[i]); ab[i] = p; }
  }
}
__global__ void k_visit(const int* __restrict__ seqs, const float* __restrict__ stime,
                        const float* __restrict__ emb, const float* __restrict__ time_w,
                        const float* __restrict__ time_b, const float* __restrict__ tup_w,
                        const float* __restrict__ tup_b, float* __restrict__ ve,
                        float* __restrict__ tave, float* __restrict__ taveF) {
  __shared__ float tf[64];
  int bt = blockIdx.x;
  int b = bt / VV, t = bt % VV;
  int j = threadIdx.x;
  float ts = stime[bt] / 180.f;
  if (j < 64) {
    float x = ts * time_w[j] + time_b[j];
    tf[j] = 1.f - tanhf(x * x);
  }
  __syncthreads();
  float s = 0.f;
  for (int c = 0; c < CCODES; ++c) {
    int code = seqs[bt * CCODES + c];
    float e = emb[code * DD + j];
    s += (e > 0.f) ? e : 0.f;
  }
  float te = tup_b[j];
#pragma unroll 4
  for (int u = 0; u < 64; ++u) te += tf[u] * tup_w[u * DD + j];
  float tv = s + te;
  ve[bt * DD + j] = s;
  tave[bt * DD + j] = tv;
  taveF[(b * VV + (VV - 1 - t)) * DD + j] = tv;
}
__global__ void k_logits(const float* __restrict__ Lh, const float* __restrict__ Rh,
                         const float* __restrict__ w, const float* __restrict__ bsc,
                         float* __restrict__ logits) {
  int i = blockIdx.x * 256 + threadIdx.x;
  if (i >= BB * VV) return;
  float s = bsc[0];
  const float* lp = Lh + i * HH;
  const float* rp = Rh + i * HH;
  for (int c = 0; c < HH; ++c) s += lp[c] * w[c] + rp[c] * w[HH + c];
  logits[i] = s;
}
__global__ void k_attn_sel(const float* __restrict__ logits, int* __restrict__ selmask,
                           int* __restrict__ idxb) {
  int b = threadIdx.x;
  if (b >= BB) return;
  float a[VV]; int ord[VV];
  float mx = -1e30f;
  for (int t = 0; t < VV; ++t) { float v = logits[b * VV + t]; a[t] = v; if (v > mx) mx = v; }
  float ssum = 0.f;
  for (int t = 0; t < VV; ++t) { a[t] = __expf(a[t] - mx); ssum += a[t]; }
  for (int t = 0; t < VV; ++t) { a[t] /= ssum; ord[t] = t; }
  for (int i = 0; i < VV; ++i) {
    int best = i;
    for (int k = i + 1; k < VV; ++k) if (a[ord[k]] > a[ord[best]]) best = k;
    int tmp = ord[i]; ord[i] = ord[best]; ord[best] = tmp;
  }
  float cum = 0.f;
  for (int i = 0; i < VV; ++i) { selmask[b * VV + ord[i]] = (cum < 0.9f) ? 1 : 0; cum += a[ord[i]]; }
  for (int k = 0; k < KTOP; ++k) idxb[b * KTOP + k] = ord[k];
}
__global__ void k_tvn(const float* __restrict__ ve, const int* __restrict__ idxb,
                      float* __restrict__ tvn) {
  int i = blockIdx.x * 256 + threadIdx.x;
  if (i >= BB * KTOP * DD) return;
  int c = i & 255, k = (i >> 8) % KTOP, b = i / (KTOP * DD);
  tvn[i] = ve[(b * VV + idxb[b * KTOP + k]) * DD + c] + 0.05f * gaussk(0, 0, i);
}
__global__ void k_temb(float* __restrict__ temb, int g, int sid) {
  int b = blockIdx.x, j = threadIdx.x;  // block 128
  int t = tstep(g, sid, b);
  int jj = j & 63;
  float f = __expf(-logf(10000.f) * jj / 64.f);
  float e = (float)t * f;
  temb[b * 128 + j] = (j < 64) ? sinf(e) : cosf(e);
}
__global__ void k_diffin1(const float* __restrict__ tvn, const float* __restrict__ ab,
                          const float* __restrict__ tproj, float* __restrict__ u, int gp) {
  int i = blockIdx.x * 256 + threadIdx.x;
  if (i >= BB * KTOP * DD) return;
  int b = i / (KTOP * DD);
  float a = ab[tstep(gp, 1, b)];
  float n1 = gaussk(gp, 2, i);
  u[i] = tvn[i] * sqrtf(a) + n1 * sqrtf(1.f - a) + tproj[(b << 8) + (i & 255)];
}
__global__ void k_gen_scatter(const float* __restrict__ tvn, const float* __restrict__ dnout,
                              const int* __restrict__ idxb, float* __restrict__ temp, int gp) {
  int b = blockIdx.x, c = threadIdx.x;
  for (int k = 0; k < KTOP; ++k) {
    int i = (b * KTOP + k) * DD + c;
    float val = tvn[i] + gaussk(gp, 2, i) - dnout[i];
    temp[(b * VV + idxb[b * KTOP + k]) * DD + c] = val;
  }
}
__global__ void k_diffin2(const float* __restrict__ hog, const float* __restrict__ ab,
                          const float* __restrict__ tproj, float* __restrict__ u, int gp) {
  int i = blockIdx.x * 256 + threadIdx.x;
  if (i >= BB * VV * DD) return;
  int b = i / (VV * DD);
  float a = ab[tstep(gp, 3, b)];
  float n2 = gaussk(gp, 4, i);
  u[i] = hog[i] * sqrtf(a) + n2 * sqrtf(1.f - a) + tproj[(b << 8) + (i & 255)];
}
__global__ void k_newv(const float* __restrict__ hog, const float* __restrict__ dnout,
                       const int* __restrict__ selmask, const int* __restrict__ idxb,
                       float* __restrict__ temp, int gp) {
  int i = blockIdx.x * 256 + threadIdx.x;
  if (i >= BB * VV * DD) return;
  int b = i / (VV * DD);
  int t = (i >> 8) % VV;
  if (selmask[b * VV + t] && t != idxb[b * KTOP]) {
    temp[i] = hog[i] + gaussk(gp, 4, i) - dnout[i];
  }
}
__global__ void k_clfin(const float* __restrict__ Lh, const float* __restrict__ Rh,
                        float* __restrict__ clfin) {
  int i = blockIdx.x * 256 + threadIdx.x;
  if (i >= BB * 2 * HH) return;
  int b = i >> 9, c = i & 511;
  clfin[i] = (c < HH) ? Lh[(b * VV + VV - 1) * HH + c]
                      : Rh[(b * VV + VV - 1) * HH + (c - HH)];
}
__global__ void k_c3(const float* __restrict__ clf2, const float* __restrict__ w,
                     const float* __restrict__ bias, float* __restrict__ preds, int slot) {
  int o = threadIdx.x;
  if (o >= BB * 2) return;
  int b = o >> 1, j = o & 1;
  float s = bias[j];
  for (int c = 0; c < 1024; ++c) s += clf2[b * 1024 + c] * w[c * 2 + j];
  preds[b * (2 * (GGEN + 1)) + slot * 2 + j] = s;
}
__global__ void k_vote(const float* __restrict__ preds, const float* __restrict__ w,
                       const float* __restrict__ bias, float* __restrict__ outp) {
  int o = threadIdx.x;
  if (o >= BB * 2) return;
  int b = o >> 1, j = o & 1;
  float s = bias[j];
  for (int i = 0; i < 2 * (GGEN + 1); ++i) s += preds[b * 12 + i] * w[i * 2 + j];
  outp[b * 2 + j] = s;
}

// ---------------------------------------------------------------------------
// host launcher
// ---------------------------------------------------------------------------
extern "C" void kernel_launch(void* const* d_in, const int* in_sizes, int n_in,
                              void* d_out, int out_size, void* d_ws, size_t ws_size,
                              hipStream_t stream) {
  (void)in_sizes; (void)n_in; (void)out_size; (void)ws_size;
  enum {
    IN_SEQS = 0, IN_TIME, P_EMB, P_TIME_W, P_TIME_B, P_TIMEUP_W, P_TIMEUP_B,
    P_L_WI0, P_L_WH0, P_L_B0, P_L_WI1, P_L_WH1, P_L_B1,
    P_R_WI0, P_R_WH0, P_R_B0, P_R_WI1, P_R_WH1, P_R_B1,
    P_VATTN_W, P_VATTN_B, P_AGG_W, P_AGG_B,
    P_C1_W, P_C1_B, P_C2_W, P_C2_B, P_C3_W, P_C3_B,
    P_VOTE_W, P_VOTE_B, P_DT_W, P_DT_B, P_D1_W, P_D1_B, P_D2_W, P_D2_B,
    IN_BETAS
  };
  const int* seqs = (const int*)d_in[IN_SEQS];
  const float* stime = (const float*)d_in[IN_TIME];
  auto F = [&](int i) { return (const float*)d_in[i]; };
  float* outp = (float*)d_out;

  // bump arena in workspace
  char* wsp = (char*)d_ws;
  auto alloc = [&](size_t bytes) {
    void* p = (void*)wsp;
    wsp += (bytes + 255) & ~(size_t)255;
    return p;
  };
  const size_t NE = (size_t)BB * VV * DD;  // 819200
  auto f32buf = [&](size_t n) { return (float*)alloc(n * sizeof(float)); };
  auto f16buf = [&](size_t n) { return (_Float16*)alloc(n * sizeof(_Float16)); };

  // f16 transposed weight copies (Wt[n*K + k])
  _Float16* wi0L = f16buf(DD * FH); _Float16* wh0L = f16buf(HH * FH);
  _Float16* wi1L = f16buf(HH * FH); _Float16* wh1L = f16buf(HH * FH);
  _Float16* wi0R = f16buf(DD * FH); _Float16* wh0R = f16buf(HH * FH);
  _Float16* wi1R = f16buf(HH * FH); _Float16* wh1R = f16buf(HH * FH);
  _Float16* c1w = f16buf(512 * 2048);
  _Float16* c2w = f16buf(2048 * 1024);
  _Float16* aggw = f16buf(768 * 256);
  _Float16* d1w = f16buf(256 * 512);
  _Float16* d2w = f16buf(512 * 256);
  _Float16* dtw = f16buf(128 * 256);

  // f32 buffers
  float* ve = f32buf(NE);
  float* tave = f32buf(NE);
  float* flipb = f32buf(NE);
  float* temp = f32buf(NE);
  float* preA = f32buf((size_t)BB * VV * FH);
  float* preB = f32buf((size_t)BB * VV * FH);
  float* hA0 = f32buf(NE); float* hB0 = f32buf(NE);
  float* hA1 = f32buf(NE); float* hB1 = f32buf(NE);
  float* hog = f32buf(NE);
  float* dn2 = f32buf(NE);
  float* dh = f32buf((size_t)BB * VV * 512);
  float* ubuf = f32buf(NE);
  float* tvn = f32buf((size_t)BB * KTOP * DD);
  float* tproj = f32buf((size_t)BB * DD);
  float* temb = f32buf((size_t)BB * 128);
  float* clfin = f32buf((size_t)BB * 512);
  float* clf1 = f32buf((size_t)BB * 2048);
  float* clf2 = f32buf((size_t)BB * 1024);
  float* logits = f32buf(BB * VV);
  float* preds = f32buf((size_t)BB * 2 * (GGEN + 1));
  float* abar = f32buf(TDIFF);
  int* selmask = (int*)alloc(BB * VV * sizeof(int));
  int* idxb = (int*)alloc(BB * KTOP * sizeof(int));

  auto cvtT = [&](int pi, _Float16* dst, int K, int N) {
    k_cvtT<<<(K * N + 255) / 256, 256, 0, stream>>>(F(pi), dst, K, N);
  };
  auto gemm = [&](const float* A, const _Float16* Wt, const float* bias, float* C,
                  int M, int N, int K, int ldw, int relu, int accm, int ash, int cm) {
    k_gemm<<<dim3(N / 64, M / 64), 256, 0, stream>>>(A, Wt, bias, C, M, N, K, ldw,
                                                     relu, accm, ash, cm);
  };
  auto lstm2_pair = [&](const float* x, const float* xF, float* oL, float* oR) {
    gemm(x, wi0L, F(P_L_B0), preA, BB * VV, FH, DD, DD, 0, 0, 0, 1);
    gemm(xF, wi0R, F(P_R_B0), preB, BB * VV, FH, DD, DD, 0, 0, 0, 1);
    k_scan<<<2, 1024, 0, stream>>>(preA, preB, wh0L, wh0R, hA0, hB0);
    gemm(hA0, wi1L, F(P_L_B1), preA, BB * VV, FH, HH, HH, 0, 0, 0, 1);
    gemm(hB0, wi1R, F(P_R_B1), preB, BB * VV, FH, HH, HH, 0, 0, 0, 1);
    k_scan<<<2, 1024, 0, stream>>>(preA, preB, wh1L, wh1R, oL, oR);
  };
  auto classify = [&](int slot) {
    k_clfin<<<(BB * 512 + 255) / 256, 256, 0, stream>>>(hA1, hB1, clfin);
    gemm(clfin, c1w, F(P_C1_B), clf1, BB, 2048, 512, 512, 1, 0, 0, 0);
    gemm(clf1, c2w, F(P_C2_B), clf2, BB, 1024, 2048, 2048, 1, 0, 0, 0);
    k_c3<<<1, 128, 0, stream>>>(clf2, F(P_C3_W), F(P_C3_B), preds, slot);
  };

  // ---- weights -> transposed f16 ----
  cvtT(P_L_WI0, wi0L, DD, FH); cvtT(P_L_WH0, wh0L, HH, FH);
  cvtT(P_L_WI1, wi1L, HH, FH); cvtT(P_L_WH1, wh1L, HH, FH);
  cvtT(P_R_WI0, wi0R, DD, FH); cvtT(P_R_WH0, wh0R, HH, FH);
  cvtT(P_R_WI1, wi1R, HH, FH); cvtT(P_R_WH1, wh1R, HH, FH);
  cvtT(P_C1_W, c1w, 512, 2048); cvtT(P_C2_W, c2w, 2048, 1024);
  cvtT(P_AGG_W, aggw, 768, 256);
  cvtT(P_D1_W, d1w, 256, 512); cvtT(P_D2_W, d2w, 512, 256);
  cvtT(P_DT_W, dtw, 128, 256);
  k_alphabar<<<1, 1, 0, stream>>>(F(IN_BETAS), abar);

  // ---- embeddings ----
  k_visit<<<BB * VV, 256, 0, stream>>>(seqs, stime, F(P_EMB), F(P_TIME_W), F(P_TIME_B),
                                       F(P_TIMEUP_W), F(P_TIMEUP_B), ve, tave, flipb);

  // ---- initial bidirectional LSTM + attention + pred[0] ----
  lstm2_pair(tave, flipb, hA1, hB1);
  k_logits<<<(BB * VV + 255) / 256, 256, 0, stream>>>(hA1, hB1, F(P_VATTN_W),
                                                      F(P_VATTN_B), logits);
  k_attn_sel<<<1, 64, 0, stream>>>(logits, selmask, idxb);
  classify(0);
  k_tvn<<<(BB * KTOP * DD + 255) / 256, 256, 0, stream>>>(ve, idxb, tvn);

  // ---- generation loop ----
  for (int g = 0; g < GGEN; ++g) {
    int gp = g + 1;
    // diffusion pass 1 on top-k visits
    k_temb<<<BB, 128, 0, stream>>>(temb, gp, 1);
    gemm(temb, dtw, F(P_DT_B), tproj, BB, DD, 128, 128, 0, 0, 0, 0);
    k_diffin1<<<(BB * KTOP * DD + 255) / 256, 256, 0, stream>>>(tvn, abar, tproj, ubuf, gp);
    gemm(ubuf, d1w, F(P_D1_B), dh, BB * KTOP, 512, 256, 256, 1, 0, 0, 0);
    gemm(dh, d2w, F(P_D2_B), dn2, BB * KTOP, 256, 512, 512, 0, 0, 0, 0);
    k_copy<<<(int)((NE + 255) / 256), 256, 0, stream>>>(tave, temp, (int)NE);
    k_gen_scatter<<<BB, 256, 0, stream>>>(tvn, dn2, idxb, temp, gp);
    // recompute directional states on augmented sequence
    k_flip<<<(int)((NE + 255) / 256), 256, 0, stream>>>(temp, flipb);
    lstm2_pair(temp, flipb, hA1, hB1);  // Lh2, Rh2
    // hog = [temp, L_prev, R_next] @ agg_w + agg_b  (K chunks of transposed agg)
    gemm(temp, aggw, F(P_AGG_B), hog, BB * VV, DD, DD, 768, 0, 0, 0, 0);
    gemm(hA1, aggw + 256, nullptr, hog, BB * VV, DD, DD, 768, 0, 1, -1, 0);
    gemm(hB1, aggw + 512, nullptr, hog, BB * VV, DD, DD, 768, 0, 1, +1, 0);
    // diffusion pass 2 on all selected visits
    k_temb<<<BB, 128, 0, stream>>>(temb, gp, 3);
    gemm(temb, dtw, F(P_DT_B), tproj, BB, DD, 128, 128, 0, 0, 0, 0);
    k_diffin2<<<(int)((NE + 255) / 256), 256, 0, stream>>>(hog, abar, tproj, ubuf, gp);
    gemm(ubuf, d1w, F(P_D1_B), dh, BB * VV, 512, 256, 256, 1, 0, 0, 0);
    gemm(dh, d2w, F(P_D2_B), dn2, BB * VV, 256, 512, 512, 0, 0, 0, 0);
    k_newv<<<(int)((NE + 255) / 256), 256, 0, stream>>>(hog, dn2, selmask, idxb, temp, gp);
    // final states + prediction for this generation
    k_flip<<<(int)((NE + 255) / 256), 256, 0, stream>>>(temp, flipb);
    lstm2_pair(temp, flipb, hA1, hB1);  // Lh3, Rh3
    classify(g + 1);
  }
  k_vote<<<1, 128, 0, stream>>>(preds, F(P_VOTE_W), F(P_VOTE_B), outp);
}
